// MixtureOfAggregators_81329500717374
// MI455X (gfx1250) — matrix operand order repaired and checked by reading, not compiled, not run on hardware
//
#include <hip/hip_runtime.h>
#include <math.h>

typedef __bf16 bf16;
typedef __attribute__((ext_vector_type(16))) __bf16 v16bf;
typedef __attribute__((ext_vector_type(8)))  __bf16 v8bf;
typedef __attribute__((ext_vector_type(8)))  float  v8f;

constexpr int Bc = 8, Nc = 1024, INc = 2048, DIMc = 512, Hc = 8, DHc = 64,
              DEPTHc = 2, Ec = 4, Cc = 10;
constexpr int Sc = Nc + 1;        // 1025 (cls + tokens)
constexpr int ROWS = Bc * Sc;     // 8200

__device__ __forceinline__ float gelu_f(float x) {
  const float k0 = 0.7978845608028654f, k1 = 0.044715f;
  return 0.5f * x * (1.f + tanhf(k0 * (x + k1 * x * x * x)));
}

__device__ __forceinline__ v8bf pack8(const float4& a, const float4& b) {
  v8bf p;
  p[0] = (bf16)a.x; p[1] = (bf16)a.y; p[2] = (bf16)a.z; p[3] = (bf16)a.w;
  p[4] = (bf16)b.x; p[5] = (bf16)b.y; p[6] = (bf16)b.z; p[7] = (bf16)b.w;
  return p;
}

// ---------------------------------------------------------------------------
// Generic GEMM: C[M,N] = act(A[M,K] @ Bm[K,N] + bias) (+ resid)
// 64x128 block tile, BK=32, 8 waves, double-buffered LDS, bf16 WMMA f32 acc.
// Each wave: 16 rows x 64 cols -> 1 A-frag reused over 4 B-frags = 4 WMMA/step.
// OOB rows handled by clamping (outputs for those rows are never stored).
// Requires: N % 128 == 0, K % 32 == 0.
// ---------------------------------------------------------------------------
__global__ __launch_bounds__(256) void gemm_bf16_k(
    const float* __restrict__ A, const float* __restrict__ Bm,
    const float* __restrict__ bias, const float* __restrict__ resid,
    float* __restrict__ C, int M, int N, int K, int act) {
  __shared__ __align__(16) bf16 As[2][64 * 40];    // [row][k]
  __shared__ __align__(16) bf16 Bs[2][128 * 40];   // [n][k] (transposed)
  const int tiles_n = N >> 7;
  const int bm = blockIdx.x / tiles_n, bn = blockIdx.x % tiles_n;
  const int tid = threadIdx.x, lane = tid & 31, wv = tid >> 5;
  const int half = lane >> 4, lm = lane & 15;
  const int wm = (wv & 3) << 4;   // wave row offset in block tile
  const int wn = (wv >> 2) << 6;  // wave col offset (0 or 64)
  v8f acc[4] = {{}, {}, {}, {}};
  const int arow = tid >> 2, akb = (tid & 3) << 3;  // A: 64 rows x 32 k
  const int brow = tid >> 3, bnb = (tid & 7) << 4;  // B: 32 k-rows x 128 n
  const int grc = min(bm * 64 + arow, M - 1);       // clamped (no branch)
  const float* __restrict__ aptr = A + (size_t)grc * K + akb;
  const float* __restrict__ bptr = Bm + (size_t)brow * N + bn * 128 + bnb;

  float4 ar0, ar1, br[4];
  auto gload = [&](int k0) {
    ar0 = *(const float4*)(aptr + k0);
    ar1 = *(const float4*)(aptr + k0 + 4);
    const float* bp = bptr + (size_t)k0 * N;
    br[0] = *(const float4*)(bp);
    br[1] = *(const float4*)(bp + 4);
    br[2] = *(const float4*)(bp + 8);
    br[3] = *(const float4*)(bp + 12);
  };
  auto sstore = [&](int buf) {
    *(v8bf*)(&As[buf][arow * 40 + akb]) = pack8(ar0, ar1);
#pragma unroll
    for (int q = 0; q < 4; ++q) {
      Bs[buf][(bnb + q * 4 + 0) * 40 + brow] = (bf16)br[q].x;
      Bs[buf][(bnb + q * 4 + 1) * 40 + brow] = (bf16)br[q].y;
      Bs[buf][(bnb + q * 4 + 2) * 40 + brow] = (bf16)br[q].z;
      Bs[buf][(bnb + q * 4 + 3) * 40 + brow] = (bf16)br[q].w;
    }
  };

  gload(0);
  sstore(0);
  __syncthreads();
  const int nk = K >> 5;
  for (int t = 0; t < nk; ++t) {
    const int buf = t & 1;
    if (t + 1 < nk) gload((t + 1) << 5);  // keep loads in flight over WMMA
    // A fragment (16x32): lane<16 K 0-7/16-23, lane>=16 K 8-15/24-31
    v8bf t0, t1;
    v16bf af;
    t0 = *(const v8bf*)(&As[buf][(wm + lm) * 40 + half * 8]);
    t1 = *(const v8bf*)(&As[buf][(wm + lm) * 40 + 16 + half * 8]);
#pragma unroll
    for (int j = 0; j < 8; ++j) { af[j] = t0[j]; af[j + 8] = t1[j]; }
    // B fragments (32x16): lane<16 K 0-7/16-23, lane>=16 K 8-15/24-31
#pragma unroll
    for (int c = 0; c < 4; ++c) {
      v16bf bfr;
      t0 = *(const v8bf*)(&Bs[buf][(wn + c * 16 + lm) * 40 + half * 8]);
      t1 = *(const v8bf*)(&Bs[buf][(wn + c * 16 + lm) * 40 + 16 + half * 8]);
#pragma unroll
      for (int j = 0; j < 8; ++j) { bfr[j] = t0[j]; bfr[j + 8] = t1[j]; }
      acc[c] = __builtin_amdgcn_wmma_f32_16x16x32_bf16(
          false, af, false, bfr, (short)0, acc[c], false, false);
    }
    if (t + 1 < nk) sstore(buf ^ 1);
    __syncthreads();
  }
  // Epilogue: lane holds col lm (+16c), rows (8*half + i)
  const int rbase = bm * 64 + wm + half * 8;
  const int cb0 = bn * 128 + wn + lm;
  float bz[4];
#pragma unroll
  for (int c = 0; c < 4; ++c) bz[c] = bias ? bias[cb0 + c * 16] : 0.f;
#pragma unroll
  for (int i = 0; i < 8; ++i) {
    const int r = rbase + i;
    if (r < M) {
#pragma unroll
      for (int c = 0; c < 4; ++c) {
        float v = acc[c][i] + bz[c];
        if (act == 1) v = fmaxf(v, 0.f);
        else if (act == 2) v = gelu_f(v);
        if (resid) v += resid[(size_t)r * N + cb0 + c * 16];
        C[(size_t)r * N + cb0 + c * 16] = v;
      }
    }
  }
}

// ---------------------------------------------------------------------------
// Flash attention: block = (b, h, 128 q rows); kv tiles of 32, double-buffered.
// Each wave owns 16 q rows -> softmax stats stay inside the wave.
// OOB kv rows are clamped to row Sc-1 (finite data); their scores are masked
// to -1e30 so exp()=0, hence V contributions vanish exactly.
// ---------------------------------------------------------------------------
__global__ __launch_bounds__(256) void attn_k(const float* __restrict__ QKV,
                                              float* __restrict__ O) {
  __shared__ __align__(16) bf16 Qs[128 * 72];     // [row][dh]
  __shared__ __align__(16) bf16 Ks[2][32 * 72];   // [kv][dh]
  __shared__ __align__(16) bf16 Vs[2][64 * 40];   // [dh][kv] (transposed)
  __shared__ __align__(16) bf16 Ps[128 * 40];     // [row][kv]
  const int qblocks = (Sc + 127) >> 7;  // 9
  const int qb = blockIdx.x % qblocks;
  const int h = (blockIdx.x / qblocks) % Hc;
  const int b = blockIdx.x / (qblocks * Hc);
  const int tid = threadIdx.x, lane = tid & 31, wv = tid >> 5;
  const int half = lane >> 4, lm = lane & 15;
  const float scale = 0.125f;  // DH^-0.5
  {  // load + scale Q block (clamped rows; OOB rows never stored)
    const int row = tid >> 1, cb = (tid & 1) * 32;
    const int s = min(qb * 128 + row, Sc - 1);
    const float* qp = QKV + (size_t)(b * Sc + s) * 1536 + h * 64 + cb;
#pragma unroll
    for (int j = 0; j < 4; ++j) {
      float4 q0 = *(const float4*)(qp + j * 8);
      float4 q1 = *(const float4*)(qp + j * 8 + 4);
      q0.x *= scale; q0.y *= scale; q0.z *= scale; q0.w *= scale;
      q1.x *= scale; q1.y *= scale; q1.z *= scale; q1.w *= scale;
      *(v8bf*)(&Qs[row * 72 + cb + j * 8]) = pack8(q0, q1);
    }
  }
  // K/V tile staging (double buffered)
  const int kvr = tid >> 3, cbk = (tid & 7) << 3;
  float4 kr0, kr1, vr0, vr1;
  auto kvload = [&](int kv0) {
    const int s = min(kv0 + kvr, Sc - 1);
    const float* kp = QKV + (size_t)(b * Sc + s) * 1536 + 512 + h * 64 + cbk;
    kr0 = *(const float4*)(kp);
    kr1 = *(const float4*)(kp + 4);
    vr0 = *(const float4*)(kp + 512);
    vr1 = *(const float4*)(kp + 516);
  };
  auto kvstore = [&](int buf) {
    *(v8bf*)(&Ks[buf][kvr * 72 + cbk]) = pack8(kr0, kr1);
    Vs[buf][(cbk + 0) * 40 + kvr] = (bf16)vr0.x;
    Vs[buf][(cbk + 1) * 40 + kvr] = (bf16)vr0.y;
    Vs[buf][(cbk + 2) * 40 + kvr] = (bf16)vr0.z;
    Vs[buf][(cbk + 3) * 40 + kvr] = (bf16)vr0.w;
    Vs[buf][(cbk + 4) * 40 + kvr] = (bf16)vr1.x;
    Vs[buf][(cbk + 5) * 40 + kvr] = (bf16)vr1.y;
    Vs[buf][(cbk + 6) * 40 + kvr] = (bf16)vr1.z;
    Vs[buf][(cbk + 7) * 40 + kvr] = (bf16)vr1.w;
  };
  kvload(0);
  kvstore(0);
  __syncthreads();

  v16bf aq0, aq1;  // Q A-fragments, two K-steps over DH=64
  {
    v8bf t0, t1;
    t0 = *(const v8bf*)(&Qs[(wv * 16 + lm) * 72 + half * 8]);
    t1 = *(const v8bf*)(&Qs[(wv * 16 + lm) * 72 + 16 + half * 8]);
#pragma unroll
    for (int j = 0; j < 8; ++j) { aq0[j] = t0[j]; aq0[j + 8] = t1[j]; }
    t0 = *(const v8bf*)(&Qs[(wv * 16 + lm) * 72 + 32 + half * 8]);
    t1 = *(const v8bf*)(&Qs[(wv * 16 + lm) * 72 + 48 + half * 8]);
#pragma unroll
    for (int j = 0; j < 8; ++j) { aq1[j] = t0[j]; aq1[j + 8] = t1[j]; }
  }
  v8f oacc[4] = {{}, {}, {}, {}};
  float mst[8], lst[8];
#pragma unroll
  for (int i = 0; i < 8; ++i) { mst[i] = -1e30f; lst[i] = 0.f; }

  const int ntiles = (Sc + 31) >> 5;  // 33
  for (int t = 0; t < ntiles; ++t) {
    const int buf = t & 1;
    const int kv0 = t << 5;
    if (t + 1 < ntiles) kvload((t + 1) << 5);  // prefetch next K/V tile
    // scores S = Q @ K^T : two 16x16 fragments (kv cols 0-15, 16-31)
    v8f s0 = {}, s1 = {};
    {
      v16bf bk; v8bf t0v, t1v;
#pragma unroll
      for (int ks = 0; ks < 2; ++ks) {
        t0v = *(const v8bf*)(&Ks[buf][lm * 72 + ks * 32 + half * 8]);
        t1v = *(const v8bf*)(&Ks[buf][lm * 72 + ks * 32 + 16 + half * 8]);
#pragma unroll
        for (int j = 0; j < 8; ++j) { bk[j] = t0v[j]; bk[j + 8] = t1v[j]; }
        s0 = __builtin_amdgcn_wmma_f32_16x16x32_bf16(false, ks ? aq1 : aq0,
                false, bk, (short)0, s0, false, false);
        t0v = *(const v8bf*)(&Ks[buf][(16 + lm) * 72 + ks * 32 + half * 8]);
        t1v = *(const v8bf*)(&Ks[buf][(16 + lm) * 72 + ks * 32 + 16 + half * 8]);
#pragma unroll
        for (int j = 0; j < 8; ++j) { bk[j] = t0v[j]; bk[j + 8] = t1v[j]; }
        s1 = __builtin_amdgcn_wmma_f32_16x16x32_bf16(false, ks ? aq1 : aq0,
                false, bk, (short)0, s1, false, false);
      }
    }
    const bool inv0 = (kv0 + lm) >= Sc;
    const bool inv1 = (kv0 + 16 + lm) >= Sc;
    float p0[8], p1[8];
#pragma unroll
    for (int i = 0; i < 8; ++i) {
      float a = inv0 ? -1e30f : s0[i];
      float c = inv1 ? -1e30f : s1[i];
      float mx = fmaxf(a, c);
      mx = fmaxf(mx, __shfl_xor(mx, 1));
      mx = fmaxf(mx, __shfl_xor(mx, 2));
      mx = fmaxf(mx, __shfl_xor(mx, 4));
      mx = fmaxf(mx, __shfl_xor(mx, 8));
      const float mnew = fmaxf(mst[i], mx);
      const float al = __expf(mst[i] - mnew);
      const float e0 = __expf(a - mnew);
      const float e1 = __expf(c - mnew);
      float sm = e0 + e1;
      sm += __shfl_xor(sm, 1); sm += __shfl_xor(sm, 2);
      sm += __shfl_xor(sm, 4); sm += __shfl_xor(sm, 8);
      lst[i] = lst[i] * al + sm;
      mst[i] = mnew;
      p0[i] = e0; p1[i] = e1;
      oacc[0][i] *= al; oacc[1][i] *= al; oacc[2][i] *= al; oacc[3][i] *= al;
    }
    // stage P (bf16) through this wave's private LDS strip, reload as A-frag
#pragma unroll
    for (int i = 0; i < 8; ++i) {
      const int pr = wv * 16 + half * 8 + i;
      Ps[pr * 40 + lm] = (bf16)p0[i];
      Ps[pr * 40 + 16 + lm] = (bf16)p1[i];
    }
    asm volatile("s_wait_dscnt 0x0" ::: "memory");  // same-wave LDS RAW
    {
      v16bf ap, bv; v8bf t0v, t1v;
      t0v = *(const v8bf*)(&Ps[(wv * 16 + lm) * 40 + half * 8]);
      t1v = *(const v8bf*)(&Ps[(wv * 16 + lm) * 40 + 16 + half * 8]);
#pragma unroll
      for (int j = 0; j < 8; ++j) { ap[j] = t0v[j]; ap[j + 8] = t1v[j]; }
#pragma unroll
      for (int c = 0; c < 4; ++c) {
        t0v = *(const v8bf*)(&Vs[buf][(c * 16 + lm) * 40 + half * 8]);
        t1v = *(const v8bf*)(&Vs[buf][(c * 16 + lm) * 40 + 16 + half * 8]);
#pragma unroll
        for (int j = 0; j < 8; ++j) { bv[j] = t0v[j]; bv[j + 8] = t1v[j]; }
        oacc[c] = __builtin_amdgcn_wmma_f32_16x16x32_bf16(
            false, ap, false, bv, (short)0, oacc[c], false, false);
      }
    }
    if (t + 1 < ntiles) kvstore(buf ^ 1);
    __syncthreads();
  }
#pragma unroll
  for (int i = 0; i < 8; ++i) {
    const int s = qb * 128 + wv * 16 + half * 8 + i;
    if (s < Sc) {
      const float inv = lst[i] > 0.f ? 1.f / lst[i] : 0.f;
      const size_t base = (size_t)(b * Sc + s) * 512 + h * 64;
      O[base + 0 + lm] = oacc[0][i] * inv;
      O[base + 16 + lm] = oacc[1][i] * inv;
      O[base + 32 + lm] = oacc[2][i] * inv;
      O[base + 48 + lm] = oacc[3][i] * inv;
    }
  }
}

// ---------------------------------------------------------------------------
// LayerNorm over DIM=512; one block per row; strides allow cls-row pooling.
// ---------------------------------------------------------------------------
__global__ __launch_bounds__(256) void ln_k(const float* __restrict__ in,
                                            const float* __restrict__ g,
                                            const float* __restrict__ bb,
                                            float* __restrict__ out,
                                            long in_stride, long out_stride) {
  __shared__ float r1[256], r2[256];
  const long row = blockIdx.x;
  const float* x = in + row * in_stride;
  const int t = threadIdx.x;
  const float v0 = x[t], v1 = x[t + 256];
  r1[t] = v0 + v1; r2[t] = v0 * v0 + v1 * v1;
  __syncthreads();
  for (int off = 128; off > 0; off >>= 1) {
    if (t < off) { r1[t] += r1[t + off]; r2[t] += r2[t + off]; }
    __syncthreads();
  }
  const float mean = r1[0] * (1.f / 512.f);
  const float var = r2[0] * (1.f / 512.f) - mean * mean;
  const float rstd = rsqrtf(var + 1e-5f);
  float* o = out + row * out_stride;
  o[t] = (v0 - mean) * rstd * g[t] + bb[t];
  o[t + 256] = (v1 - mean) * rstd * g[t + 256] + bb[t + 256];
}

__global__ void pool_k(const float* __restrict__ in, float* __restrict__ out) {
  const int idx = blockIdx.x * blockDim.x + threadIdx.x;  // 4096
  const int b = idx >> 9, d = idx & 511;
  const float* p = in + (size_t)b * Nc * DIMc + d;
  float s = 0.f;
  for (int n = 0; n < Nc; ++n) s += p[(size_t)n * DIMc];
  out[idx] = s * (1.f / Nc);
}

__global__ __launch_bounds__(256) void router_k(const float* __restrict__ rin,
                                                const float* __restrict__ Wrf,
                                                const float* __restrict__ brf,
                                                float* __restrict__ gout) {
  const int t = threadIdx.x, lane = t & 31, b = t >> 5;  // 8 waves, one per b
  float acc[Ec] = {0.f, 0.f, 0.f, 0.f};
  for (int d = lane; d < DIMc; d += 32) {
    const float xv = rin[b * DIMc + d];
#pragma unroll
    for (int e = 0; e < Ec; ++e) acc[e] += xv * Wrf[d * Ec + e];
  }
#pragma unroll
  for (int e = 0; e < Ec; ++e)
    for (int off = 16; off > 0; off >>= 1) acc[e] += __shfl_xor(acc[e], off);
  if (lane == 0) {
    float mx = -1e30f;
    for (int e = 0; e < Ec; ++e) { acc[e] += brf[e]; mx = fmaxf(mx, acc[e]); }
    float s = 0.f, ex[Ec];
    for (int e = 0; e < Ec; ++e) { ex[e] = __expf(acc[e] - mx); s += ex[e]; }
    for (int e = 0; e < Ec; ++e) gout[b * Ec + e] = ex[e] / s;
  }
}

__global__ void assemble_k(const float* __restrict__ P,
                           const float* __restrict__ cls,
                           float* __restrict__ T) {
  const int idx = blockIdx.x * blockDim.x + threadIdx.x;
  if (idx >= Bc * Sc * DIMc) return;
  const int d = idx & 511;
  const int s = (idx >> 9) % Sc;
  const int b = idx / (Sc * DIMc);
  T[idx] = (s == 0) ? cls[d] : P[((size_t)b * Nc + (s - 1)) * DIMc + d];
}

__global__ void head_k(const float* __restrict__ lat,
                       const float* __restrict__ Wh,
                       const float* __restrict__ bh,
                       float* __restrict__ logi) {
  const int idx = blockIdx.x * blockDim.x + threadIdx.x;
  if (idx >= Bc * Cc) return;
  const int b = idx / Cc, c = idx % Cc;
  float s = bh[c];
  const float* l = lat + (size_t)b * DIMc;
  for (int d = 0; d < DIMc; ++d) s += l[d] * Wh[d * Cc + c];
  logi[idx] = s;
}

__global__ void combine_k(const float* __restrict__ latE,
                          const float* __restrict__ logE,
                          const float* __restrict__ g,
                          float* __restrict__ out) {
  const int idx = blockIdx.x * blockDim.x + threadIdx.x;
  if (idx < Bc * DIMc) {
    const int b = idx >> 9, d = idx & 511;
    float s = 0.f;
    for (int e = 0; e < Ec; ++e)
      s += g[b * Ec + e] * latE[((size_t)e * Bc + b) * DIMc + d];
    out[idx] = s;
  } else if (idx < Bc * DIMc + Bc * Cc) {
    const int k = idx - Bc * DIMc;
    const int b = k / Cc, c = k % Cc;
    float s = 0.f;
    for (int e = 0; e < Ec; ++e)
      s += g[b * Ec + e] * logE[(e * Bc + b) * Cc + c];
    out[idx] = s;
  }
}

extern "C" void kernel_launch(void* const* d_in, const int* in_sizes, int n_in,
                              void* d_out, int out_size, void* d_ws,
                              size_t ws_size, hipStream_t stream) {
  (void)in_sizes; (void)n_in; (void)out_size; (void)ws_size;
  const float* x    = (const float*)d_in[0];
  const float* Wr   = (const float*)d_in[1];
  const float* br   = (const float*)d_in[2];
  const float* Wrf  = (const float*)d_in[3];
  const float* brf  = (const float*)d_in[4];
  const float* Wp   = (const float*)d_in[5];
  const float* bp   = (const float*)d_in[6];
  const float* cls  = (const float*)d_in[7];
  const float* ln1g = (const float*)d_in[8];
  const float* ln1b = (const float*)d_in[9];
  const float* Wqkv = (const float*)d_in[10];
  const float* bqkv = (const float*)d_in[11];
  const float* Wo   = (const float*)d_in[12];
  const float* bo   = (const float*)d_in[13];
  const float* ln2g = (const float*)d_in[14];
  const float* ln2b = (const float*)d_in[15];
  const float* W1   = (const float*)d_in[16];
  const float* b1   = (const float*)d_in[17];
  const float* W2   = (const float*)d_in[18];
  const float* b2   = (const float*)d_in[19];
  const float* lnfg = (const float*)d_in[20];
  const float* lnfb = (const float*)d_in[21];
  const float* Wh   = (const float*)d_in[22];
  const float* bh   = (const float*)d_in[23];
  float* out = (float*)d_out;
  float* ws = (float*)d_ws;
  float* buf0 = ws;                              // 8200*512 (proj / FFN hidden)
  float* T    = buf0 + (size_t)ROWS * DIMc;      // 8200*512
  float* Hb   = T + (size_t)ROWS * DIMc;         // 8200*512
  float* QKV  = Hb + (size_t)ROWS * DIMc;        // 8200*1536
  float* Ob   = QKV + (size_t)ROWS * 3 * DIMc;   // 8200*512
  float* rin  = Ob + (size_t)ROWS * DIMc;        // 8*512
  float* latE = rin + Bc * DIMc;                 // 4*8*512
  float* logE = latE + Ec * Bc * DIMc;           // 4*8*10
  float* gsoft = out + Bc * DIMc + Bc * Cc;      // third output, in d_out

  const int MB = Bc * Nc;                 // 8192
  const int mt = (ROWS + 63) / 64;        // 129 row tiles for S-length GEMMs

  // Router: relu(x@Wr+br) -> mean pool -> softmax(rin@Wrf+brf)
  gemm_bf16_k<<<(MB / 64) * (DIMc / 128), 256, 0, stream>>>(
      x, Wr, br, nullptr, buf0, MB, DIMc, INc, 1);
  pool_k<<<16, 256, 0, stream>>>(buf0, rin);
  router_k<<<1, 256, 0, stream>>>(rin, Wrf, brf, gsoft);

  for (int e = 0; e < Ec; ++e) {
    gemm_bf16_k<<<(MB / 64) * (DIMc / 128), 256, 0, stream>>>(
        x, Wp + (size_t)e * INc * DIMc, bp + e * DIMc, nullptr, buf0,
        MB, DIMc, INc, 1);
    assemble_k<<<(Bc * Sc * DIMc + 255) / 256, 256, 0, stream>>>(
        buf0, cls + e * DIMc, T);
    for (int l = 0; l < DEPTHc; ++l) {
      const int el = e * DEPTHc + l;
      ln_k<<<ROWS, 256, 0, stream>>>(T, ln1g + (size_t)el * DIMc,
                                     ln1b + (size_t)el * DIMc, Hb, DIMc, DIMc);
      gemm_bf16_k<<<mt * (3 * DIMc / 128), 256, 0, stream>>>(
          Hb, Wqkv + (size_t)el * DIMc * 3 * DIMc, bqkv + (size_t)el * 3 * DIMc,
          nullptr, QKV, ROWS, 3 * DIMc, DIMc, 0);
      attn_k<<<Bc * Hc * ((Sc + 127) / 128), 256, 0, stream>>>(QKV, Ob);
      gemm_bf16_k<<<mt * (DIMc / 128), 256, 0, stream>>>(
          Ob, Wo + (size_t)el * DIMc * DIMc, bo + (size_t)el * DIMc, T, T,
          ROWS, DIMc, DIMc, 0);
      ln_k<<<ROWS, 256, 0, stream>>>(T, ln2g + (size_t)el * DIMc,
                                     ln2b + (size_t)el * DIMc, Hb, DIMc, DIMc);
      gemm_bf16_k<<<mt * (DIMc / 128), 256, 0, stream>>>(
          Hb, W1 + (size_t)el * DIMc * DIMc, b1 + (size_t)el * DIMc, nullptr,
          buf0, ROWS, DIMc, DIMc, 2);
      gemm_bf16_k<<<mt * (DIMc / 128), 256, 0, stream>>>(
          buf0, W2 + (size_t)el * DIMc * DIMc, b2 + (size_t)el * DIMc, T, T,
          ROWS, DIMc, DIMc, 0);
    }
    // Final LN over cls rows (stride Sc*DIM picks t[:,0]) then local head.
    ln_k<<<Bc, 256, 0, stream>>>(T, lnfg + (size_t)e * DIMc,
                                 lnfb + (size_t)e * DIMc,
                                 latE + (size_t)e * Bc * DIMc,
                                 (long)Sc * DIMc, DIMc);
    head_k<<<1, 128, 0, stream>>>(latE + (size_t)e * Bc * DIMc,
                                  Wh + (size_t)e * DIMc * Cc, bh + e * Cc,
                                  logE + e * Bc * Cc);
  }
  combine_k<<<(Bc * DIMc + Bc * Cc + 255) / 256, 256, 0, stream>>>(
      latE, logE, gsoft, out);
}
